// TransformerDecoderLayer_7825430414103
// MI455X (gfx1250) — compile-verified, matrix operand order checked by reading, not run on hardware
//
#include <hip/hip_runtime.h>

// ---------------- problem constants ----------------
#define D_    256
#define H_    8
#define HD_   32
#define DFF_  2048
#define NQ_   300
#define BS_   16
#define HW_   1444
#define NTQ_  (NQ_ * BS_)   // 4800 tokens
#define NTM_  (HW_ * BS_)   // 23104 tokens
#define RS_   (BS_ * D_)    // 4096: stride between successive q/k indices in token layout
#define NKP_SA 320          // 300 rounded up to 32
#define NKP_CA 1472         // 1444 rounded up to 32

typedef __attribute__((ext_vector_type(16))) __bf16 v16bf;
typedef __attribute__((ext_vector_type(8)))  __bf16 v8bf;
typedef __attribute__((ext_vector_type(8)))  float  v8f;

__device__ __forceinline__ __bf16 f2bf(float f) {
  unsigned u = __builtin_bit_cast(unsigned, f);
  unsigned r = u + 0x7FFFu + ((u >> 16) & 1u);   // round-to-nearest-even
  unsigned short h = (unsigned short)(r >> 16);
  return __builtin_bit_cast(__bf16, h);
}

__device__ __forceinline__ v16bf combine8(v8bf lo, v8bf hi) {
  v16bf a;
#pragma unroll
  for (int i = 0; i < 8; i++) { a[i] = lo[i]; a[i + 8] = hi[i]; }
  return a;
}

__device__ __forceinline__ v8f vzero8() {
  v8f z;
#pragma unroll
  for (int i = 0; i < 8; i++) z[i] = 0.f;
  return z;
}

// ---------------- f32 -> bf16 convert (optional scale) ----------------
__global__ void k_cvt(const float* __restrict__ in, __bf16* __restrict__ out,
                      float scale, long n) {
  long i = (long)blockIdx.x * blockDim.x + threadIdx.x;
  if (i < n) out[i] = f2bf(in[i] * scale);
}

// ---------------- GEMM: C[M,N] = A[M,K] * W[N,K]^T + bias  ----------------
// one wave per 16(M) x 64(N) tile; bf16 WMMA, f32 accumulate.
template <bool ACCUM, bool RELU>
__global__ __launch_bounds__(32) void k_gemm(
    const __bf16* __restrict__ A, const __bf16* __restrict__ W,
    const float* __restrict__ bias, float* __restrict__ C,
    int M, int N, int K) {
  const int lane = threadIdx.x;
  const int half = lane >> 4, r = lane & 15;
  const int m0 = blockIdx.x * 16, n0 = blockIdx.y * 64;

  v8f acc[4];
#pragma unroll
  for (int j = 0; j < 4; j++) acc[j] = vzero8();

  const __bf16* arow  = A + (size_t)(m0 + r) * K;
  const __bf16* wbase = W + (size_t)n0 * K;

  for (int k0 = 0; k0 < K; k0 += 32) {
    v16bf a = combine8(*(const v8bf*)(arow + k0 + half * 8),
                       *(const v8bf*)(arow + k0 + 16 + half * 8));
#pragma unroll
    for (int j = 0; j < 4; j++) {
      const __bf16* wrow = wbase + (size_t)(j * 16 + r) * K;
      v16bf b = combine8(*(const v8bf*)(wrow + k0 + half * 8),
                         *(const v8bf*)(wrow + k0 + 16 + half * 8));
      acc[j] = __builtin_amdgcn_wmma_f32_16x16x32_bf16(
          false, a, false, b, (short)0, acc[j], false, false);
    }
  }

#pragma unroll
  for (int j = 0; j < 4; j++) {
    int col = n0 + j * 16 + r;
    float bv = bias ? bias[col] : 0.f;
#pragma unroll
    for (int i = 0; i < 8; i++) {
      int row = m0 + i + 8 * half;
      float v = acc[j][i] + bv;
      if (ACCUM) v += C[(size_t)row * N + col];
      if (RELU)  v = fmaxf(v, 0.f);
      C[(size_t)row * N + col] = v;
    }
  }
}

// ---------------- V transpose+convert: vt[b*H+h][d][k] = V[k,b,h,d] ----------------
__global__ void k_vt(const float* __restrict__ v, __bf16* __restrict__ vt,
                     int nk, int nkp) {
  long i = (long)blockIdx.x * blockDim.x + threadIdx.x;
  long total = (long)BS_ * H_ * HD_ * nkp;
  if (i >= total) return;
  int  k    = (int)(i % nkp);
  long rest = i / nkp;
  int  d    = (int)(rest % HD_);
  int  bh   = (int)(rest / HD_);
  int  b = bh / H_, h = bh % H_;
  float val = 0.f;
  if (k < nk) val = v[((size_t)k * BS_ + b) * D_ + (size_t)h * HD_ + d];
  vt[i] = f2bf(val);
}

// ---------------- flash attention: one wave per (16-query tile, b*h) ----------------
// NSRC==2: logits = q1*k1^T + q2*k2^T (DETR concat trick). Q already scaled.
template <int NSRC>
__global__ __launch_bounds__(32) void k_attn(
    const __bf16* __restrict__ q1, const __bf16* __restrict__ q2,
    const __bf16* __restrict__ k1, const __bf16* __restrict__ k2,
    const __bf16* __restrict__ vt, const unsigned char* __restrict__ mask,
    float* __restrict__ out, int nq, int nk, int nkp) {
  const int lane = threadIdx.x;
  const int half = lane >> 4, r = lane & 15;
  const int bh = blockIdx.y, b = bh / H_, h = bh % H_;
  const size_t base = (size_t)b * D_ + (size_t)h * HD_;
  const int q0 = blockIdx.x * 16;

  const __bf16* qs[2] = { q1, q2 };
  const __bf16* ks[2] = { k1, k2 };

  int qr = q0 + r; if (qr >= nq) qr = nq - 1;
  v16bf qa[NSRC];
#pragma unroll
  for (int s = 0; s < NSRC; s++) {
    const __bf16* p = qs[s] + base + (size_t)qr * RS_;
    qa[s] = combine8(*(const v8bf*)(p + half * 8),
                     *(const v8bf*)(p + 16 + half * 8));
  }

  __shared__ float Sld[16][33];
  __shared__ __align__(16) __bf16 Pld[16][32];
  __shared__ float fsc[16];
  __shared__ float lsum[16];

  v8f acc0 = vzero8(), acc1 = vzero8();
  float m_r = -__builtin_inff(), l_r = 0.f;

  const __bf16* vtb = vt + (size_t)bh * HD_ * nkp;

  for (int kk0 = 0; kk0 < nk; kk0 += 32) {
    // ---- logits for this 16q x 32k chunk ----
    v8f s0 = vzero8(), s1 = vzero8();
#pragma unroll
    for (int s = 0; s < NSRC; s++) {
      int key0 = kk0 + r;      if (key0 >= nk) key0 = nk - 1;
      int key1 = kk0 + 16 + r; if (key1 >= nk) key1 = nk - 1;
      const __bf16* p0 = ks[s] + base + (size_t)key0 * RS_;
      const __bf16* p1 = ks[s] + base + (size_t)key1 * RS_;
      v16bf b0 = combine8(*(const v8bf*)(p0 + half * 8),
                          *(const v8bf*)(p0 + 16 + half * 8));
      v16bf b1 = combine8(*(const v8bf*)(p1 + half * 8),
                          *(const v8bf*)(p1 + 16 + half * 8));
      s0 = __builtin_amdgcn_wmma_f32_16x16x32_bf16(false, qa[s], false, b0, (short)0, s0, false, false);
      s1 = __builtin_amdgcn_wmma_f32_16x16x32_bf16(false, qa[s], false, b1, (short)0, s1, false, false);
    }
#pragma unroll
    for (int i = 0; i < 8; i++) {
      Sld[i + 8 * half][r]      = s0[i];
      Sld[i + 8 * half][16 + r] = s1[i];
    }
    __syncthreads();

    // ---- online softmax (one lane per query row) ----
    if (lane < 16) {
      const int row = lane;
      float sv[32], cmax = -__builtin_inff();
#pragma unroll
      for (int c = 0; c < 32; c++) {
        int key = kk0 + c;
        float x = -__builtin_inff();
        if (key < nk && !(mask && mask[(size_t)b * nk + key])) x = Sld[row][c];
        sv[c] = x;
        cmax = fmaxf(cmax, x);
      }
      float newm = fmaxf(m_r, cmax);
      float fs = __expf(m_r - newm);
      float ps = 0.f;
#pragma unroll
      for (int c = 0; c < 32; c++) {
        float pv = __expf(sv[c] - newm);
        Pld[row][c] = f2bf(pv);
        ps += pv;
      }
      l_r = l_r * fs + ps;
      m_r = newm;
      fsc[row] = fs;
    }
    __syncthreads();

    // ---- P * V : rescale accumulators, two WMMAs (out dims 0-15, 16-31) ----
    v16bf pa = combine8(*(const v8bf*)&Pld[r][half * 8],
                        *(const v8bf*)&Pld[r][16 + half * 8]);
    const __bf16* v0 = vtb + (size_t)r * nkp + kk0;
    const __bf16* v1 = vtb + (size_t)(16 + r) * nkp + kk0;
    v16bf vb0 = combine8(*(const v8bf*)(v0 + half * 8),
                         *(const v8bf*)(v0 + 16 + half * 8));
    v16bf vb1 = combine8(*(const v8bf*)(v1 + half * 8),
                         *(const v8bf*)(v1 + 16 + half * 8));
#pragma unroll
    for (int i = 0; i < 8; i++) {
      float f = fsc[i + 8 * half];
      acc0[i] *= f; acc1[i] *= f;
    }
    acc0 = __builtin_amdgcn_wmma_f32_16x16x32_bf16(false, pa, false, vb0, (short)0, acc0, false, false);
    acc1 = __builtin_amdgcn_wmma_f32_16x16x32_bf16(false, pa, false, vb1, (short)0, acc1, false, false);
    __syncthreads();
  }

  if (lane < 16) lsum[lane] = l_r;
  __syncthreads();
#pragma unroll
  for (int i = 0; i < 8; i++) {
    int row = i + 8 * half;
    int q = q0 + row;
    if (q < nq) {
      float inv = 1.f / lsum[row];
      out[base + (size_t)q * RS_ + r]      = acc0[i] * inv;
      out[base + (size_t)q * RS_ + 16 + r] = acc1[i] * inv;
    }
  }
}

// ---------------- residual + LayerNorm (wave per token), dual f32/bf16 out --------
__global__ __launch_bounds__(32) void k_ln(
    const float* __restrict__ x, const float* __restrict__ res,
    const float* __restrict__ g, const float* __restrict__ be,
    float* __restrict__ yf, __bf16* __restrict__ yb, int ntok) {
  int t = blockIdx.x;
  if (t >= ntok) return;
  int lane = threadIdx.x;
  const float* xr = x + (size_t)t * D_;
  const float* rr = res + (size_t)t * D_;
  float v[8]; float s = 0.f;
#pragma unroll
  for (int i = 0; i < 8; i++) { int c = lane + i * 32; v[i] = xr[c] + rr[c]; s += v[i]; }
#pragma unroll
  for (int m = 16; m >= 1; m >>= 1) s += __shfl_xor(s, m, 32);
  float mean = s * (1.f / 256.f);
  float var = 0.f;
#pragma unroll
  for (int i = 0; i < 8; i++) { float d = v[i] - mean; var += d * d; }
#pragma unroll
  for (int m = 16; m >= 1; m >>= 1) var += __shfl_xor(var, m, 32);
  float rs = rsqrtf(var * (1.f / 256.f) + 1e-5f);
#pragma unroll
  for (int i = 0; i < 8; i++) {
    int c = lane + i * 32;
    float y = (v[i] - mean) * rs * g[c] + be[c];
    yf[(size_t)t * D_ + c] = y;
    if (yb) yb[(size_t)t * D_ + c] = f2bf(y);
  }
}

// ================================ host ================================
extern "C" void kernel_launch(void* const* d_in, const int* in_sizes, int n_in,
                              void* d_out, int out_size, void* d_ws, size_t ws_size,
                              hipStream_t stream) {
  (void)in_sizes; (void)n_in; (void)out_size; (void)ws_size;

  const float* tgt   = (const float*)d_in[0];
  const float* mem   = (const float*)d_in[1];
  const float* pos   = (const float*)d_in[2];
  const float* qpos  = (const float*)d_in[3];
  const float* qsine = (const float*)d_in[4];
  const unsigned char* kpm = (const unsigned char*)d_in[5];  // (BS, HW) bool

  const float* W_sa_qc = (const float*)d_in[6];  const float* b_sa_qc = (const float*)d_in[7];
  const float* W_sa_qp = (const float*)d_in[8];  const float* b_sa_qp = (const float*)d_in[9];
  const float* W_sa_kc = (const float*)d_in[10]; const float* b_sa_kc = (const float*)d_in[11];
  const float* W_sa_kp = (const float*)d_in[12]; const float* b_sa_kp = (const float*)d_in[13];
  const float* W_sa_v  = (const float*)d_in[14]; const float* b_sa_v  = (const float*)d_in[15];
  const float* W_sa_o  = (const float*)d_in[16]; const float* b_sa_o  = (const float*)d_in[17];
  const float* W_ca_qc = (const float*)d_in[18]; const float* b_ca_qc = (const float*)d_in[19];
  const float* W_ca_kc = (const float*)d_in[20]; const float* b_ca_kc = (const float*)d_in[21];
  const float* W_ca_kp = (const float*)d_in[22]; const float* b_ca_kp = (const float*)d_in[23];
  const float* W_ca_v  = (const float*)d_in[24]; const float* b_ca_v  = (const float*)d_in[25];
  const float* W_ca_si = (const float*)d_in[26]; const float* b_ca_si = (const float*)d_in[27];
  const float* W_ca_o  = (const float*)d_in[28]; const float* b_ca_o  = (const float*)d_in[29];
  const float* W_ff1   = (const float*)d_in[30]; const float* b_ff1   = (const float*)d_in[31];
  const float* W_ff2   = (const float*)d_in[32]; const float* b_ff2   = (const float*)d_in[33];
  const float* g_n1 = (const float*)d_in[34]; const float* be_n1 = (const float*)d_in[35];
  const float* g_n2 = (const float*)d_in[36]; const float* be_n2 = (const float*)d_in[37];
  const float* g_n3 = (const float*)d_in[38]; const float* be_n3 = (const float*)d_in[39];

  // ---------- workspace carve ----------
  char* ws = (char*)d_ws;
  size_t off = 0;
  auto alloc = [&](size_t bytes) -> void* {
    void* p = ws + off;
    off += (bytes + 255) & ~(size_t)255;
    return p;
  };
  const size_t SQ = (size_t)NTQ_ * D_;   // 1,228,800
  const size_t SM = (size_t)NTM_ * D_;   // 5,914,624
  const size_t SF = (size_t)NTQ_ * DFF_; // 9,830,400

  float* fbig = (float*)alloc(SM * 4);   // shared f32 temp for memory-sized projections
  float* ftmp = (float*)alloc(SQ * 4);   // shared f32 temp for query-sized tensors
  float* t1   = (float*)alloc(SQ * 4);
  float* t2   = (float*)alloc(SQ * 4);
  float* ffh  = (float*)alloc(SF * 4);

  __bf16* tgtb  = (__bf16*)alloc(SQ * 2);
  __bf16* qpb   = (__bf16*)alloc(SQ * 2);
  __bf16* qseb  = (__bf16*)alloc(SQ * 2);
  __bf16* memb  = (__bf16*)alloc(SM * 2);
  __bf16* posb  = (__bf16*)alloc(SM * 2);
  __bf16* qb    = (__bf16*)alloc(SQ * 2);
  __bf16* kb    = (__bf16*)alloc(SQ * 2);
  __bf16* vtsa  = (__bf16*)alloc((size_t)BS_ * H_ * HD_ * NKP_SA * 2);
  __bf16* kcb   = (__bf16*)alloc(SM * 2);
  __bf16* kpb   = (__bf16*)alloc(SM * 2);
  __bf16* vtca  = (__bf16*)alloc((size_t)BS_ * H_ * HD_ * NKP_CA * 2);
  __bf16* sab   = (__bf16*)alloc(SQ * 2);
  __bf16* qcb   = (__bf16*)alloc(SQ * 2);
  __bf16* sineb = (__bf16*)alloc(SQ * 2);
  __bf16* cab   = (__bf16*)alloc(SQ * 2);
  __bf16* t1b   = (__bf16*)alloc(SQ * 2);
  __bf16* t2b   = (__bf16*)alloc(SQ * 2);
  __bf16* ffhb  = (__bf16*)alloc(SF * 2);

  const size_t SW = (size_t)D_ * D_;       // 65536
  const size_t SWF = (size_t)D_ * DFF_;    // 524288
  __bf16* wb_saqc = (__bf16*)alloc(SW * 2);
  __bf16* wb_saqp = (__bf16*)alloc(SW * 2);
  __bf16* wb_sakc = (__bf16*)alloc(SW * 2);
  __bf16* wb_sakp = (__bf16*)alloc(SW * 2);
  __bf16* wb_sav  = (__bf16*)alloc(SW * 2);
  __bf16* wb_sao  = (__bf16*)alloc(SW * 2);
  __bf16* wb_caqc = (__bf16*)alloc(SW * 2);
  __bf16* wb_cakc = (__bf16*)alloc(SW * 2);
  __bf16* wb_cakp = (__bf16*)alloc(SW * 2);
  __bf16* wb_cav  = (__bf16*)alloc(SW * 2);
  __bf16* wb_casi = (__bf16*)alloc(SW * 2);
  __bf16* wb_cao  = (__bf16*)alloc(SW * 2);
  __bf16* wb_ff1  = (__bf16*)alloc(SWF * 2);
  __bf16* wb_ff2  = (__bf16*)alloc(SWF * 2);

  auto cvt = [&](const float* in, __bf16* outp, float sc, long n) {
    k_cvt<<<(unsigned)((n + 255) / 256), 256, 0, stream>>>(in, outp, sc, n);
  };

  // ---------- convert weights + inputs to bf16 ----------
  cvt(W_sa_qc, wb_saqc, 1.f, SW);  cvt(W_sa_qp, wb_saqp, 1.f, SW);
  cvt(W_sa_kc, wb_sakc, 1.f, SW);  cvt(W_sa_kp, wb_sakp, 1.f, SW);
  cvt(W_sa_v,  wb_sav,  1.f, SW);  cvt(W_sa_o,  wb_sao,  1.f, SW);
  cvt(W_ca_qc, wb_caqc, 1.f, SW);  cvt(W_ca_kc, wb_cakc, 1.f, SW);
  cvt(W_ca_kp, wb_cakp, 1.f, SW);  cvt(W_ca_v,  wb_cav,  1.f, SW);
  cvt(W_ca_si, wb_casi, 1.f, SW);  cvt(W_ca_o,  wb_cao,  1.f, SW);
  cvt(W_ff1,   wb_ff1,  1.f, SWF); cvt(W_ff2,   wb_ff2,  1.f, SWF);
  cvt(tgt,   tgtb, 1.f, SQ);
  cvt(qpos,  qpb,  1.f, SQ);
  cvt(qsine, qseb, 1.f, SQ);
  cvt(mem,   memb, 1.f, SM);
  cvt(pos,   posb, 1.f, SM);

  const dim3 gQ(NTQ_ / 16, D_ / 64);      // 300 x 4
  const dim3 gM(NTM_ / 16, D_ / 64);      // 1444 x 4
  const dim3 gF1(NTQ_ / 16, DFF_ / 64);   // 300 x 32
  const dim3 gA((NQ_ + 15) / 16, BS_ * H_); // 19 x 128
  const float SC_SA = 0.17677669529663687f; // 32^-0.5
  const float SC_CA = 0.125f;               // 64^-0.5

  // ---------- self-attention ----------
  k_gemm<false, false><<<gQ, 32, 0, stream>>>(tgtb, wb_saqc, b_sa_qc, ftmp, NTQ_, D_, D_);
  k_gemm<true,  false><<<gQ, 32, 0, stream>>>(qpb,  wb_saqp, b_sa_qp, ftmp, NTQ_, D_, D_);
  cvt(ftmp, qb, SC_SA, SQ);
  k_gemm<false, false><<<gQ, 32, 0, stream>>>(tgtb, wb_sakc, b_sa_kc, ftmp, NTQ_, D_, D_);
  k_gemm<true,  false><<<gQ, 32, 0, stream>>>(qpb,  wb_sakp, b_sa_kp, ftmp, NTQ_, D_, D_);
  cvt(ftmp, kb, 1.f, SQ);
  k_gemm<false, false><<<gQ, 32, 0, stream>>>(tgtb, wb_sav, b_sa_v, ftmp, NTQ_, D_, D_);
  {
    long n = (long)BS_ * H_ * HD_ * NKP_SA;
    k_vt<<<(unsigned)((n + 255) / 256), 256, 0, stream>>>(ftmp, vtsa, NQ_, NKP_SA);
  }
  k_attn<1><<<gA, 32, 0, stream>>>(qb, nullptr, kb, nullptr, vtsa, nullptr,
                                   ftmp, NQ_, NQ_, NKP_SA);
  cvt(ftmp, sab, 1.f, SQ);
  k_gemm<false, false><<<gQ, 32, 0, stream>>>(sab, wb_sao, b_sa_o, ftmp, NTQ_, D_, D_);
  k_ln<<<NTQ_, 32, 0, stream>>>(ftmp, tgt, g_n1, be_n1, t1, t1b, NTQ_);

  // ---------- cross-attention ----------
  k_gemm<false, false><<<gQ, 32, 0, stream>>>(t1b,  wb_caqc, b_ca_qc, ftmp, NTQ_, D_, D_);
  cvt(ftmp, qcb, SC_CA, SQ);
  k_gemm<false, false><<<gQ, 32, 0, stream>>>(qseb, wb_casi, b_ca_si, ftmp, NTQ_, D_, D_);
  cvt(ftmp, sineb, SC_CA, SQ);
  k_gemm<false, false><<<gM, 32, 0, stream>>>(memb, wb_cakc, b_ca_kc, fbig, NTM_, D_, D_);
  cvt(fbig, kcb, 1.f, SM);
  k_gemm<false, false><<<gM, 32, 0, stream>>>(posb, wb_cakp, b_ca_kp, fbig, NTM_, D_, D_);
  cvt(fbig, kpb, 1.f, SM);
  k_gemm<false, false><<<gM, 32, 0, stream>>>(memb, wb_cav, b_ca_v, fbig, NTM_, D_, D_);
  {
    long n = (long)BS_ * H_ * HD_ * NKP_CA;
    k_vt<<<(unsigned)((n + 255) / 256), 256, 0, stream>>>(fbig, vtca, HW_, NKP_CA);
  }
  k_attn<2><<<gA, 32, 0, stream>>>(qcb, sineb, kcb, kpb, vtca, kpm,
                                   ftmp, NQ_, HW_, NKP_CA);
  cvt(ftmp, cab, 1.f, SQ);
  k_gemm<false, false><<<gQ, 32, 0, stream>>>(cab, wb_cao, b_ca_o, ftmp, NTQ_, D_, D_);
  k_ln<<<NTQ_, 32, 0, stream>>>(ftmp, t1, g_n2, be_n2, t2, t2b, NTQ_);

  // ---------- FFN ----------
  k_gemm<false, true><<<gF1, 32, 0, stream>>>(t2b, wb_ff1, b_ff1, ffh, NTQ_, DFF_, D_);
  cvt(ffh, ffhb, 1.f, SF);
  k_gemm<false, false><<<gQ, 32, 0, stream>>>(ffhb, wb_ff2, b_ff2, ftmp, NTQ_, D_, DFF_);
  k_ln<<<NTQ_, 32, 0, stream>>>(ftmp, t2, g_n3, be_n3, (float*)d_out, nullptr, NTQ_);
}
